// ScaledDotProductAttention_16827681865851
// MI455X (gfx1250) — matrix operand order repaired and checked by reading, not compiled, not run on hardware
//
#include <hip/hip_runtime.h>
#include <stdint.h>

// Problem constants (fixed by the reference).
constexpr int Bn = 8;      // batch
constexpr int QL = 2048;   // query length
constexpr int KL = 2048;   // key length
constexpr int MD = 2048;   // model dim (reduction dim of projections)
constexpr int DK = 2048;   // q/k projection dim
constexpr int DV = 2048;   // v projection dim

typedef __attribute__((ext_vector_type(16))) __bf16 v16bf;
typedef __attribute__((ext_vector_type(8)))  float  v8f;

struct alignas(16) F128 { float    f[4]; };
struct alignas(16) U128 { uint32_t w[4]; };

// ---------------- CDNA5 feature detection -----------------------------------
#if defined(__has_builtin)
#  if __has_builtin(__builtin_amdgcn_global_load_async_to_lds_b128)
#    define HAVE_ASYNC_LDS 1
#  else
#    define HAVE_ASYNC_LDS 0
#  endif
#  if __has_builtin(__builtin_amdgcn_s_wait_asynccnt)
#    define HAVE_WAIT_ASYNC_BUILTIN 1
#  else
#    define HAVE_WAIT_ASYNC_BUILTIN 0
#  endif
#  if __has_builtin(__builtin_amdgcn_tensor_load_to_lds) && \
      __has_builtin(__builtin_amdgcn_s_wait_tensorcnt)
#    define HAVE_TDM 1
#  else
#    define HAVE_TDM 0
#  endif
#  if __has_builtin(__builtin_amdgcn_cvt_pk_bf16_f32)
#    define HAVE_CVT_PK_BF16 1
#  else
#    define HAVE_CVT_PK_BF16 0
#  endif
#else
#  define HAVE_ASYNC_LDS 0
#  define HAVE_WAIT_ASYNC_BUILTIN 0
#  define HAVE_TDM 0
#  define HAVE_CVT_PK_BF16 0
#endif

// Pointee type expected by the async-LDS builtins:
// 'int __attribute__((vector_size(16)))' with AS1 (global) / AS3 (LDS).
typedef __attribute__((__vector_size__(16))) int i32x4;
typedef __attribute__((address_space(1))) i32x4 gas_i32x4;
typedef __attribute__((address_space(3))) i32x4 las_i32x4;

template <int N>
__device__ __forceinline__ void wait_async() {
#if HAVE_ASYNC_LDS
#  if HAVE_WAIT_ASYNC_BUILTIN
    __builtin_amdgcn_s_wait_asynccnt((unsigned short)N);
#  else
    asm volatile("s_wait_asynccnt %0" :: "i"(N) : "memory");
#  endif
#endif
}

// Copy 16 bytes global -> LDS. Async on CDNA5, sync fallback otherwise.
__device__ __forceinline__ void copy16_g2l(uint16_t* lds_dst, const uint16_t* g_src) {
#if HAVE_ASYNC_LDS
    __builtin_amdgcn_global_load_async_to_lds_b128(
        (gas_i32x4*)g_src, (las_i32x4*)lds_dst, 0, 0);
#else
    *reinterpret_cast<U128*>(lds_dst) = *reinterpret_cast<const U128*>(g_src);
#endif
}

// ---------------- Tensor Data Mover (TDM) staging ---------------------------
#if HAVE_TDM
typedef __attribute__((__vector_size__(16))) unsigned int u32x4v;
typedef __attribute__((__vector_size__(32))) int          i32x8v;
typedef __attribute__((__vector_size__(16))) int          i32x4v;

__device__ __forceinline__ uint32_t lds_offset(const void* p) {
    return (uint32_t)(uintptr_t)(__attribute__((address_space(3))) const void*)p;
}

// DMA one [128 rows][32 bf16] tile (row stride = strideElems) into LDS.
// D# per CDNA5 ISA 8.3/8.4: 2D tensor, data_size=2B, tile 32x128.
__device__ __forceinline__ void tdm_load_tile_128x32(const uint16_t* g_src,
                                                     uint32_t lds_byte,
                                                     uint32_t strideElems) {
    const uint64_t ga = (uint64_t)(uintptr_t)g_src;
    u32x4v g0;
    g0[0] = 1u;                                                  // count=1 (valid user D#)
    g0[1] = lds_byte;                                            // lds_addr
    g0[2] = (uint32_t)(ga & 0xFFFFFFFFu);                        // global_addr[31:0]
    g0[3] = (uint32_t)((ga >> 32) & 0x01FFFFFFu) | 0x80000000u;  // addr[56:32] | type=2

    i32x8v g1;
    g1[0] = 0x00010000;                          // data_size=1 (2 bytes); no multicast
    g1[1] = (int)(2048u << 16);                  // tensor_dim0[15:0] = 2048 (bits 63:48)
    g1[2] = (int)((2048u >> 16) | (128u << 16)); // tensor_dim0 hi | tensor_dim1 = 128
    g1[3] = (int)(32u << 16);                    // tensor_dim1 hi | tile_dim0 = 32
    g1[4] = (int)128u;                           // tile_dim1 = 128, tile_dim2 = 0
    g1[5] = (int)strideElems;                    // tensor_dim0_stride[31:0]
    g1[6] = 0;                                   // stride0 hi | tensor_dim1_stride lo
    g1[7] = 0;

    const i32x4v z4 = {0, 0, 0, 0};
    const i32x8v z8 = {0, 0, 0, 0, 0, 0, 0, 0};
    __builtin_amdgcn_tensor_load_to_lds(g0, g1, z4, z4, z8, 0);
}
#endif

// ---------------------------------------------------------------------------
__device__ __forceinline__ uint16_t f32_to_bf16(float f) {
    uint32_t u = __float_as_uint(f);
    uint32_t r = u + 0x7FFFu + ((u >> 16) & 1u);   // round-to-nearest-even
    return (uint16_t)(r >> 16);
}
#if HAVE_CVT_PK_BF16
typedef __attribute__((ext_vector_type(2))) __bf16 v2bf;
union PkBf { v2bf v; uint32_t u; };
__device__ __forceinline__ uint32_t pk_bf16(float lo, float hi) {
    PkBf r; r.v = __builtin_amdgcn_cvt_pk_bf16_f32(lo, hi);
    return r.u;
}
#else
__device__ __forceinline__ uint32_t pk_bf16(float lo, float hi) {
    return (uint32_t)f32_to_bf16(lo) | ((uint32_t)f32_to_bf16(hi) << 16);
}
#endif

union FragB16 { v16bf v; uint32_t u[8]; };

// A fragment: 16x32 bf16 (lane half h holds k = 16h..16h+15). LDS tile stored [m][k], stride 32.
__device__ __forceinline__ void frag_a(FragB16& f, const uint16_t* s, int rb, int lane) {
    const int m = rb + (lane & 15);
    const int half = lane >> 4;
    #pragma unroll
    for (int j = 0; j < 8; ++j) {
        const int kk = 2 * j + 16 * half;
        f.u[j] = *reinterpret_cast<const uint32_t*>(&s[m * 32 + kk]);
    }
}
// B fragment: 32x16 bf16. LDS tile stored [n][k], stride 32.
__device__ __forceinline__ void frag_b(FragB16& f, const uint16_t* s, int cb, int lane) {
    const int n = cb + (lane & 15);
    const int half = lane >> 4;
    #pragma unroll
    for (int j = 0; j < 8; ++j) {
        const int kk = 2 * j + 16 * half;
        f.u[j] = *reinterpret_cast<const uint32_t*>(&s[n * 32 + kk]);
    }
}

#define WMMA_BF16(A, B, C) \
    __builtin_amdgcn_wmma_f32_16x16x32_bf16(false, (A).v, false, (B).v, (short)0, (C), false, false)

// ---------------------------------------------------------------------------
// Projection GEMM:  Y[row][n] = sum_m X[row][m] * W[n][m]      (NT, f32 in)
// Block: 256 threads (8 waves), block tile 128x128, wave tile 32x64, K-step 32.
// TRANS=false: Y bf16 [rows][N] row-major.
// TRANS=true : Y bf16 stored as [batch][n][k] (per-batch transposed, rows = b*2048+k).
// ---------------------------------------------------------------------------
template <bool TRANS>
__global__ __launch_bounds__(256)
void proj_gemm_bf16(const float* __restrict__ X, const float* __restrict__ W,
                    uint16_t* __restrict__ Y, int N, int Kd) {
    __shared__ uint16_t As[128 * 32];
    __shared__ uint16_t Bs[128 * 32];

    const int t = threadIdx.x;
    const int lane = t & 31, wid = t >> 5;
    const int wrow = (wid & 3) * 32, wcol = (wid >> 2) * 64;
    const int row0 = blockIdx.y * 128, col0 = blockIdx.x * 128;

    v8f c[2][4];
    #pragma unroll
    for (int i = 0; i < 2; ++i)
        #pragma unroll
        for (int j = 0; j < 4; ++j)
            #pragma unroll
            for (int e = 0; e < 8; ++e) c[i][j][e] = 0.0f;

    const int r = t >> 1, h = t & 1;           // staging row / half-row
    for (int k0 = 0; k0 < Kd; k0 += 32) {
        {
            const float* pa = X + (size_t)(row0 + r) * Kd + k0 + h * 16;
            if (k0 + 32 < Kd) __builtin_prefetch(pa + 32, 0, 1);
            uint32_t* dst = reinterpret_cast<uint32_t*>(&As[r * 32 + h * 16]);
            #pragma unroll
            for (int e = 0; e < 4; ++e) {
                F128 q = reinterpret_cast<const F128*>(pa)[e];
                dst[e * 2 + 0] = pk_bf16(q.f[0], q.f[1]);
                dst[e * 2 + 1] = pk_bf16(q.f[2], q.f[3]);
            }
        }
        {
            const float* pb = W + (size_t)(col0 + r) * Kd + k0 + h * 16;
            if (k0 + 32 < Kd) __builtin_prefetch(pb + 32, 0, 1);
            uint32_t* dst = reinterpret_cast<uint32_t*>(&Bs[r * 32 + h * 16]);
            #pragma unroll
            for (int e = 0; e < 4; ++e) {
                F128 q = reinterpret_cast<const F128*>(pb)[e];
                dst[e * 2 + 0] = pk_bf16(q.f[0], q.f[1]);
                dst[e * 2 + 1] = pk_bf16(q.f[2], q.f[3]);
            }
        }
        __syncthreads();

        FragB16 a[2], b[4];
        #pragma unroll
        for (int i = 0; i < 2; ++i) frag_a(a[i], As, wrow + i * 16, lane);
        #pragma unroll
        for (int j = 0; j < 4; ++j) frag_b(b[j], Bs, wcol + j * 16, lane);
        #pragma unroll
        for (int i = 0; i < 2; ++i)
            #pragma unroll
            for (int j = 0; j < 4; ++j) c[i][j] = WMMA_BF16(a[i], b[j], c[i][j]);
        __syncthreads();
    }

    const int half = lane >> 4, nn = lane & 15;
    #pragma unroll
    for (int ti = 0; ti < 2; ++ti)
        #pragma unroll
        for (int tj = 0; tj < 4; ++tj)
            #pragma unroll
            for (int rr = 0; rr < 8; ++rr) {
                const int m = row0 + wrow + ti * 16 + rr + 8 * half;
                const int n = col0 + wcol + tj * 16 + nn;
                const uint16_t hv = f32_to_bf16(c[ti][tj][rr]);
                if constexpr (TRANS) {
                    // Y[b][n][k] with m = b*2048 + k
                    Y[((size_t)(m >> 11) * 2048 + n) * 2048 + (m & 2047)] = hv;
                } else {
                    Y[(size_t)m * N + n] = hv;
                }
            }
}

// ---------------------------------------------------------------------------
// Unified bf16 NT GEMM, double-buffered LDS staging via TDM (preferred) or
// async-LDS copies (fallback).
//   C[m][n] = sum_k A[m][k] * B[n][k]            (both K-contiguous bf16)
// EPI=0: scores epilogue (scale + mask -> f32 S)
// EPI=1: plain f32 store (attention output)
// Block: 256 threads, block tile 128x128, K-step 32, 2-deep pipeline.
// ---------------------------------------------------------------------------
template <int EPI>
__global__ __launch_bounds__(256)
void gemm_bf16_nt(const uint16_t* __restrict__ A, const uint16_t* __restrict__ Bm,
                  float* __restrict__ C, int Kd,
                  const unsigned char* __restrict__ mask,
                  const int* __restrict__ pcausal, float scale) {
    __shared__ uint16_t As[2][128 * 32];
    __shared__ uint16_t Bs[2][128 * 32];

    const int t = threadIdx.x;
    const int lane = t & 31, wid = t >> 5;
    const int wrow = (wid & 3) * 32, wcol = (wid >> 2) * 64;
    const int row0 = blockIdx.y * 128, col0 = blockIdx.x * 128;

    v8f c[2][4];
    #pragma unroll
    for (int i = 0; i < 2; ++i)
        #pragma unroll
        for (int j = 0; j < 4; ++j)
            #pragma unroll
            for (int e = 0; e < 8; ++e) c[i][j][e] = 0.0f;

    auto compute = [&](int buf) {
        FragB16 a[2], b[4];
        #pragma unroll
        for (int i = 0; i < 2; ++i) frag_a(a[i], As[buf], wrow + i * 16, lane);
        #pragma unroll
        for (int j = 0; j < 4; ++j) frag_b(b[j], Bs[buf], wcol + j * 16, lane);
        #pragma unroll
        for (int i = 0; i < 2; ++i)
            #pragma unroll
            for (int j = 0; j < 4; ++j) c[i][j] = WMMA_BF16(a[i], b[j], c[i][j]);
    };

    const int nsteps = Kd >> 5;
    int buf = 0;

#if HAVE_TDM
    // Wave 0 drives the Tensor Data Mover; TENSORcnt is per-wave, so wave 0
    // waits and the workgroup barrier publishes the LDS tiles to all waves.
    if (wid == 0) {
        tdm_load_tile_128x32(A  + (size_t)row0 * Kd, lds_offset(&As[0][0]), (uint32_t)Kd);
        tdm_load_tile_128x32(Bm + (size_t)col0 * Kd, lds_offset(&Bs[0][0]), (uint32_t)Kd);
    }
    for (int s = 0; s < nsteps; ++s) {
        if (wid == 0) {
            if (s + 1 < nsteps) {
                const int kn = (s + 1) * 32;
                tdm_load_tile_128x32(A  + (size_t)row0 * Kd + kn,
                                     lds_offset(&As[buf ^ 1][0]), (uint32_t)Kd);
                tdm_load_tile_128x32(Bm + (size_t)col0 * Kd + kn,
                                     lds_offset(&Bs[buf ^ 1][0]), (uint32_t)Kd);
                __builtin_amdgcn_s_wait_tensorcnt((unsigned short)2);
            } else {
                __builtin_amdgcn_s_wait_tensorcnt((unsigned short)0);
            }
        }
        __syncthreads();
        compute(buf);
        __syncthreads();
        buf ^= 1;
    }
#else
    // Async-LDS path: 512 16B chunks per tile; 2 per thread per tile.
    auto stage = [&](int bb, int k0) {
        #pragma unroll
        for (int i = 0; i < 2; ++i) {
            const int ch = t + 256 * i;
            const int rr = ch >> 2, seg = ch & 3;
            copy16_g2l(&As[bb][rr * 32 + seg * 8],
                       A + (size_t)(row0 + rr) * Kd + k0 + seg * 8);
            copy16_g2l(&Bs[bb][rr * 32 + seg * 8],
                       Bm + (size_t)(col0 + rr) * Kd + k0 + seg * 8);
        }
    };
    stage(0, 0);
    for (int s = 0; s < nsteps; ++s) {
        if (s + 1 < nsteps) {
            stage(buf ^ 1, (s + 1) * 32);
            wait_async<4>();
        } else {
            wait_async<0>();
        }
        __syncthreads();
        compute(buf);
        __syncthreads();
        buf ^= 1;
    }
#endif

    const int half = lane >> 4, nn = lane & 15;
    const int causal = (EPI == 0) ? pcausal[0] : 0;
    #pragma unroll
    for (int ti = 0; ti < 2; ++ti)
        #pragma unroll
        for (int tj = 0; tj < 4; ++tj)
            #pragma unroll
            for (int rr = 0; rr < 8; ++rr) {
                const int m = row0 + wrow + ti * 16 + rr + 8 * half;
                const int n = col0 + wcol + tj * 16 + nn;
                if constexpr (EPI == 0) {
                    // Mirror reference: keep where (mask[k] || (causal && k > q)).
                    const bool keep = (mask[n] != 0) || (causal && (n > m));
                    const float s = c[ti][tj][rr] * scale;
                    C[(size_t)m * KL + n] = keep ? s : -1.0e30f;
                } else {
                    C[(size_t)m * DV + n] = c[ti][tj][rr];
                }
            }
}

// ---------------------------------------------------------------------------
// Row softmax.  P[q][k] = softmax_k(S[q][:]) as bf16.  256 threads per row.
// ---------------------------------------------------------------------------
__global__ __launch_bounds__(256)
void softmax_rows(const float* __restrict__ S, uint16_t* __restrict__ P) {
    __shared__ float red[256];
    const int q = blockIdx.x, t = threadIdx.x;
    const float* row = S + (size_t)q * KL;

    float m = -3.0e38f;
    for (int j = t; j < KL; j += 256) m = fmaxf(m, row[j]);
    red[t] = m; __syncthreads();
    #pragma unroll
    for (int s = 128; s > 0; s >>= 1) {
        if (t < s) red[t] = fmaxf(red[t], red[t + s]);
        __syncthreads();
    }
    m = red[0]; __syncthreads();

    float sum = 0.0f;
    for (int j = t; j < KL; j += 256) {
        const float sv = row[j];
        sum += (sv <= -1.0e29f) ? 0.0f : __expf(sv - m);
    }
    red[t] = sum; __syncthreads();
    #pragma unroll
    for (int s = 128; s > 0; s >>= 1) {
        if (t < s) red[t] += red[t + s];
        __syncthreads();
    }
    sum = red[0];
    const float rcp = (sum > 0.0f) ? 1.0f / sum : 0.0f;

    for (int j = t; j < KL; j += 256) {
        const float sv = row[j];
        const float p = (sv <= -1.0e29f) ? 0.0f : __expf(sv - m) * rcp;
        P[(size_t)q * KL + j] = f32_to_bf16(p);
    }
}

// ---------------------------------------------------------------------------
// Host-side launcher.
// Workspace: qb/kb (bf16 [b][q][d]), vbT (bf16 [b][d][k]), S f32, P bf16.
// ---------------------------------------------------------------------------
extern "C" void kernel_launch(void* const* d_in, const int* in_sizes, int n_in,
                              void* d_out, int out_size, void* d_ws, size_t ws_size,
                              hipStream_t stream) {
    const float* query = (const float*)d_in[0];
    const float* key_i = (const float*)d_in[1];
    const float* value = (const float*)d_in[2];
    const float* Wq    = (const float*)d_in[3];
    const float* Wk    = (const float*)d_in[4];
    const float* Wv    = (const float*)d_in[5];
    const unsigned char* mask = (const unsigned char*)d_in[6];
    const int* pcausal = (const int*)d_in[7];
    float* out = (float*)d_out;

    char* w = (char*)d_ws;
    const size_t SZ = (size_t)Bn * QL * DK * sizeof(uint16_t);   // 64 MB
    uint16_t* qb  = (uint16_t*)(w);
    uint16_t* kb  = (uint16_t*)(w + SZ);
    uint16_t* vbT = (uint16_t*)(w + 2 * SZ);                     // [b][d][k]
    float*    Sb  = (float*)   (w + 3 * SZ);
    uint16_t* Pb  = (uint16_t*)(w + 3 * SZ + (size_t)QL * KL * sizeof(float));

    const dim3 blk(256);
    const dim3 gproj(DK / 128, (Bn * QL) / 128);
    proj_gemm_bf16<false><<<gproj, blk, 0, stream>>>(query, Wq, qb,  DK, MD);
    proj_gemm_bf16<false><<<gproj, blk, 0, stream>>>(key_i, Wk, kb,  DK, MD);
    proj_gemm_bf16<true ><<<gproj, blk, 0, stream>>>(value, Wv, vbT, DV, MD);

    const float scale = 0.022097086912079608f;   // 1/sqrt(2048)
    const dim3 gs(KL / 128, QL / 128);
    const dim3 go(DV / 128, QL / 128);
    for (int b = 0; b < Bn; ++b) {
        gemm_bf16_nt<0><<<gs, blk, 0, stream>>>(qb + (size_t)b * QL * DK,
                                                kb + (size_t)b * KL * DK,
                                                Sb, DK, mask + (size_t)b * KL,
                                                pcausal, scale);
        softmax_rows<<<dim3(QL), dim3(256), 0, stream>>>(Sb, Pb);
        gemm_bf16_nt<1><<<go, blk, 0, stream>>>(Pb, vbT + (size_t)b * (size_t)DV * KL,
                                                out + (size_t)b * QL * DV, KL,
                                                nullptr, nullptr, 1.0f);
    }
}
